// WindowMSA_15513421873331
// MI455X (gfx1250) — matrix operand order repaired and checked by reading, not compiled, not run on hardware
//
#include <hip/hip_runtime.h>

#define NTOK 49   // tokens per window
#define NP   64   // padded tokens
#define CCH  192  // channels
#define NH   6    // heads
#define HD   32   // head dim

typedef __attribute__((ext_vector_type(16))) __bf16 v16bf;
typedef __attribute__((ext_vector_type(2)))  __bf16 bf16x2;
typedef __attribute__((ext_vector_type(8)))  float  v8f;
typedef __attribute__((ext_vector_type(4)))  int    v4i;

union Frag { unsigned int u[8]; v16bf v; };
union Acc  { v8f v; float f[8]; };

#define AS1 __attribute__((address_space(1)))
#define AS3 __attribute__((address_space(3)))

#if __has_builtin(__builtin_amdgcn_global_load_async_to_lds_b128)
#define USE_ASYNC_LDS 1
#else
#define USE_ASYNC_LDS 0
#endif

// pack two f32 -> packed bf16x2 (native v_cvt, RNE)
__device__ __forceinline__ unsigned int pkbf2(float lo, float hi) {
  bf16x2 t;
  t.x = (__bf16)lo;
  t.y = (__bf16)hi;
  return __builtin_bit_cast(unsigned int, t);
}

__device__ __forceinline__ void async_copy16(const void* g, void* l) {
#if USE_ASYNC_LDS
  __builtin_amdgcn_global_load_async_to_lds_b128((AS1 v4i*)g, (AS3 v4i*)l, 0, 0);
#else
  *(float4*)l = *(const float4*)g;
#endif
}

__device__ __forceinline__ void wait_async0() {
#if USE_ASYNC_LDS
#if __has_builtin(__builtin_amdgcn_s_wait_asynccnt)
  __builtin_amdgcn_s_wait_asynccnt(0);
#else
  asm volatile("s_wait_asynccnt 0" ::: "memory");
#endif
#endif
}

// A-matrix fragment (16x32 bf16, M x K), source row-major [M][K] with given stride.
// ISA layout: lane m = lane%16, half = lane/16; VGPR v holds K = (v/4)*16 + (v%4)*2 + half*8 (pair)
__device__ __forceinline__ v16bf load_a(const __bf16* base, int stride,
                                        int m0, int k0, int lane) {
  Frag f;
  const int m = m0 + (lane & 15);
  const int half = lane >> 4;
  const __bf16* row = base + m * stride + k0;
#pragma unroll
  for (int r = 0; r < 8; ++r) {
    int k = ((r >> 2) << 4) + ((r & 3) << 1) + (half << 3);
    f.u[r] = *(const unsigned int*)(row + k);
  }
  return f.v;
}

// B-matrix fragment (32x16 bf16, K x N) where B[k][n] = Bsrc[n][k], Bsrc row-major [N][K].
// ISA layout: lane n = lane%16, half = lane/16; VGPR v holds K = (v/4)*16 + (v%4)*2 + half*8 (pair)
__device__ __forceinline__ v16bf load_b(const __bf16* base, int stride,
                                        int n0, int k0, int lane) {
  Frag f;
  const int n = n0 + (lane & 15);
  const int half = lane >> 4;
  const __bf16* row = base + n * stride + k0;
#pragma unroll
  for (int r = 0; r < 8; ++r)
    f.u[r] = *(const unsigned int*)(row + ((r >> 2) << 4) + ((r & 3) << 1) + (half << 3));
  return f.v;
}

__device__ __forceinline__ v8f wmma_bf16(v16bf a, v16bf b, v8f c) {
  return __builtin_amdgcn_wmma_f32_16x16x32_bf16(false, a, false, b, (short)0, c,
                                                 false, false);
}

__global__ void cvt_weights_kernel(const float* __restrict__ Wq,
                                   const float* __restrict__ Wkv,
                                   const float* __restrict__ Wo,
                                   __bf16* __restrict__ wq,
                                   __bf16* __restrict__ wkv,
                                   __bf16* __restrict__ wo) {
  int i = blockIdx.x * blockDim.x + threadIdx.x;
  if (i < CCH * CCH) { wq[i] = (__bf16)Wq[i]; wo[i] = (__bf16)Wo[i]; }
  if (i < 2 * CCH * CCH) wkv[i] = (__bf16)Wkv[i];
}

__global__ __launch_bounds__(256)
void winmsa_kernel(const float* __restrict__ q, const float* __restrict__ kv,
                   const float* __restrict__ mask,
                   const float* __restrict__ bq, const float* __restrict__ bkv,
                   const float* __restrict__ bo, const float* __restrict__ rpb,
                   const __bf16* __restrict__ wq,
                   const __bf16* __restrict__ wkv,
                   const __bf16* __restrict__ wo,
                   float* __restrict__ out, int nW) {
  extern __shared__ char smem[];
  // Region A [0, 75264): f32 staging for async copies; dead after stage 0b,
  //                      then reused as Qh/Kh/Vt (73728 B).
  // Region B [75264, 124416): qin/kvin bf16; X aliases qin, S/P alias kvin.
  float*  stageQ = (float*)smem;                       // [49*192] f32
  float*  stageK = stageQ + NTOK * CCH;                // [49*192] f32
  __bf16* Qh     = (__bf16*)smem;                      // [64][192] bf16 (alias A)
  __bf16* Kh     = Qh + NP * CCH;                      // [64][192] bf16 (alias A)
  __bf16* Vt     = Kh + NP * CCH;                      // [192][64] bf16 (alias A)
  __bf16* qin    = (__bf16*)(smem + 2 * NTOK * CCH * 4);  // [64][192] bf16
  __bf16* kvin   = qin + NP * CCH;                     // [64][192] bf16
  __bf16* X      = qin;                                // alias (qin dead after stage 1)
  float*  S      = (float*)kvin;                       // [64][64] f32 (alias)
  __bf16* P      = kvin + NP * NP * 2;                 // [64][64] bf16 (alias)

  const int tid  = threadIdx.x;
  const int lane = tid & 31;
  const int wave = tid >> 5;
  const int b    = blockIdx.x;
  const int w    = b % nW;

  // ---- Stage 0a: async-copy q/kv windows (raw f32) into LDS staging ----
  const float* qg  = q  + (size_t)b * NTOK * CCH;
  const float* kvg = kv + (size_t)b * NTOK * CCH;
  for (int i = tid; i < NTOK * CCH / 4; i += 256) {   // 2352 x 16B chunks each
    async_copy16(qg + i * 4, stageQ + i * 4);
    async_copy16(kvg + i * 4, stageK + i * 4);
  }
  wait_async0();
  __syncthreads();

  // ---- Stage 0b: convert to bf16 (packed pairs), zero-pad rows 49..63 ----
  {
    unsigned int* q32 = (unsigned int*)qin;
    unsigned int* k32 = (unsigned int*)kvin;
    for (int i = tid; i < NP * CCH / 2; i += 256) {
      int r = i / (CCH / 2);
      float q0 = 0.f, q1 = 0.f, k0 = 0.f, k1 = 0.f;
      if (r < NTOK) {
        q0 = stageQ[2 * i]; q1 = stageQ[2 * i + 1];
        k0 = stageK[2 * i]; k1 = stageK[2 * i + 1];
      }
      q32[i] = pkbf2(q0, q1);
      k32[i] = pkbf2(k0, k1);
    }
  }
  __syncthreads();   // staging dead; Qh/Kh/Vt aliases become live

  // ---- Stage 1: Q projection  Qh = q @ Wq^T + bq  (48 tiles of 16x16, K=192) ----
  for (int t = wave; t < 48; t += 8) {
    int mt = t & 3, nt = t >> 2;
    Acc acc = {};
#pragma unroll
    for (int kt = 0; kt < 6; ++kt) {
      v16bf a  = load_a(qin, CCH, mt * 16, kt * 32, lane);
      v16bf bb = load_b(wq,  CCH, nt * 16, kt * 32, lane);
      acc.v = wmma_bf16(a, bb, acc.v);
    }
    int n = nt * 16 + (lane & 15);
    float bias = bq[n];
    int mbase = mt * 16 + ((lane >> 4) << 3);
#pragma unroll
    for (int r = 0; r < 8; ++r)
      Qh[(mbase + r) * CCH + n] = (__bf16)(acc.f[r] + bias);
  }

  // ---- Stage 2: KV projection (96 tiles, cols 0..191 -> K, 192..383 -> V transposed) ----
  for (int t = wave; t < 96; t += 8) {
    int mt = t & 3, nt = t >> 2;   // nt in 0..23
    Acc acc = {};
#pragma unroll
    for (int kt = 0; kt < 6; ++kt) {
      v16bf a  = load_a(kvin, CCH, mt * 16, kt * 32, lane);
      v16bf bb = load_b(wkv,  CCH, nt * 16, kt * 32, lane);
      acc.v = wmma_bf16(a, bb, acc.v);
    }
    int n = nt * 16 + (lane & 15);
    float bias = bkv[n];
    int mbase = mt * 16 + ((lane >> 4) << 3);
    if (nt < 12) {
#pragma unroll
      for (int r = 0; r < 8; ++r)
        Kh[(mbase + r) * CCH + n] = (__bf16)(acc.f[r] + bias);
    } else {
      int c2 = n - CCH;  // 0..191 -> head h = c2/32, d = c2%32
#pragma unroll
      for (int r = 0; r < 8; ++r)
        Vt[c2 * NP + (mbase + r)] = (__bf16)(acc.f[r] + bias);
    }
  }
  __syncthreads();   // qin/kvin now dead; aliases X/S/P become live

  // ---- Stage 3: per-head attention ----
  const float scale = 0.17677669529663687f;  // 32^-0.5
  for (int h = 0; h < NH; ++h) {
    // S = (Q @ K^T)*scale + bias + mask  (16 tiles, K=32 -> 1 WMMA each)
    for (int t = wave; t < 16; t += 8) {
      int mt = t & 3, nt = t >> 2;
      v16bf a  = load_a(Qh, CCH, mt * 16, h * HD, lane);
      v16bf bb = load_b(Kh, CCH, nt * 16, h * HD, lane);
      Acc acc = {};
      acc.v = wmma_bf16(a, bb, acc.v);
      int n = nt * 16 + (lane & 15);
      int mbase = mt * 16 + ((lane >> 4) << 3);
#pragma unroll
      for (int r = 0; r < 8; ++r) {
        int m = mbase + r;
        float sv = -1e30f;                      // padded key columns -> softmax 0
        if (n < NTOK) {
          sv = acc.f[r] * scale;
          if (m < NTOK) {
            int ci = 13 * (m / 7) + (m % 7);    // coords(m)
            int jj = 48 - n;                    // flip(1)
            int cj = 13 * (jj / 7) + (jj % 7);  // coords(48-n)
            sv += rpb[(ci + cj) * NH + h];
            sv += mask[((size_t)w * NTOK + m) * NTOK + n];
          }
        }
        S[m * NP + n] = sv;
      }
    }
    __syncthreads();

    // softmax: 4 lanes per row, shuffle reduction; P written as packed bf16x2
    {
      int row = tid >> 2, qtr = tid & 3;
      float* srow = S + row * NP + qtr * 16;
      float mx = -1e30f;
#pragma unroll
      for (int j = 0; j < 16; ++j) mx = fmaxf(mx, srow[j]);
      mx = fmaxf(mx, __shfl_xor(mx, 1));
      mx = fmaxf(mx, __shfl_xor(mx, 2));
      float sum = 0.f;
#pragma unroll
      for (int j = 0; j < 16; ++j) { float e = __expf(srow[j] - mx); srow[j] = e; sum += e; }
      sum += __shfl_xor(sum, 1);
      sum += __shfl_xor(sum, 2);
      float inv = 1.f / sum;
      unsigned int* prow = (unsigned int*)(P + row * NP + qtr * 16);
#pragma unroll
      for (int j = 0; j < 8; ++j)
        prow[j] = pkbf2(srow[2 * j] * inv, srow[2 * j + 1] * inv);
    }
    __syncthreads();

    // X[:, h*32 : h*32+32] = P @ V   (8 tiles, one per wave, K=64 -> 2 WMMA)
    {
      int mt = wave & 3, nt = wave >> 2;
      Acc acc = {};
#pragma unroll
      for (int kt = 0; kt < 2; ++kt) {
        v16bf a  = load_a(P, NP, mt * 16, kt * 32, lane);
        v16bf bb = load_b(Vt + h * HD * NP, NP, nt * 16, kt * 32, lane);
        acc.v = wmma_bf16(a, bb, acc.v);
      }
      int n = h * HD + nt * 16 + (lane & 15);
      int mbase = mt * 16 + ((lane >> 4) << 3);
#pragma unroll
      for (int r = 0; r < 8; ++r)
        X[(mbase + r) * CCH + n] = (__bf16)acc.f[r];
    }
    __syncthreads();
  }

  // ---- Stage 4: output projection  out = X @ Wo^T + bo ----
  const size_t obase = (size_t)b * NTOK * CCH;
  for (int t = wave; t < 48; t += 8) {
    int mt = t & 3, nt = t >> 2;
    Acc acc = {};
#pragma unroll
    for (int kt = 0; kt < 6; ++kt) {
      v16bf a  = load_a(X,  CCH, mt * 16, kt * 32, lane);
      v16bf bb = load_b(wo, CCH, nt * 16, kt * 32, lane);
      acc.v = wmma_bf16(a, bb, acc.v);
    }
    int n = nt * 16 + (lane & 15);
    float bias = bo[n];
    int mbase = mt * 16 + ((lane >> 4) << 3);
#pragma unroll
    for (int r = 0; r < 8; ++r) {
      int m = mbase + r;
      if (m < NTOK) out[obase + (size_t)m * CCH + n] = acc.f[r] + bias;
    }
  }
}

extern "C" void kernel_launch(void* const* d_in, const int* in_sizes, int n_in,
                              void* d_out, int out_size, void* d_ws, size_t ws_size,
                              hipStream_t stream) {
  const float* q    = (const float*)d_in[0];
  const float* kv   = (const float*)d_in[1];
  const float* mask = (const float*)d_in[2];
  const float* Wq   = (const float*)d_in[3];
  const float* bq   = (const float*)d_in[4];
  const float* Wkv  = (const float*)d_in[5];
  const float* bkv  = (const float*)d_in[6];
  const float* Wo   = (const float*)d_in[7];
  const float* bo   = (const float*)d_in[8];
  const float* rpb  = (const float*)d_in[9];

  const int B  = in_sizes[0] / (NTOK * CCH);
  const int nW = in_sizes[2] / (NTOK * NTOK);

  __bf16* wq_bf  = (__bf16*)d_ws;                 //  36864 elems
  __bf16* wkv_bf = wq_bf  + CCH * CCH;            //  73728 elems
  __bf16* wo_bf  = wkv_bf + 2 * CCH * CCH;        //  36864 elems

  cvt_weights_kernel<<<(2 * CCH * CCH + 255) / 256, 256, 0, stream>>>(
      Wq, Wkv, Wo, wq_bf, wkv_bf, wo_bf);

  // LDS: 75264 (f32 staging, reused as Qh/Kh/Vt) + 2*24576 (qin/kvin) = 124416 B
  const size_t smem_bytes = (size_t)2 * NTOK * CCH * 4 + (size_t)2 * NP * CCH * 2;
  winmsa_kernel<<<dim3(B), dim3(256), smem_bytes, stream>>>(
      q, kv, mask, bq, bkv, bo, rpb, wq_bf, wkv_bf, wo_bf, (float*)d_out, nW);
}
